// CPCUnsupersivedCriterion_45071386805143
// MI455X (gfx1250) — compile-verified, matrix operand order checked by reading, not run on hardware
//
#include <hip/hip_runtime.h>

// ---------------------------------------------------------------------------
// CPC unsupervised criterion, fused for gfx1250 (CDNA5, wave32, WMMA).
// Shapes (fixed by the reference):
//   G=8, S=140, D=256 (=D_AR=D_ENC), K=12, NEG=128, POOL=20000, w=S-K=128
// ---------------------------------------------------------------------------

typedef _Float16 v8h  __attribute__((ext_vector_type(8)));
typedef _Float16 v16h __attribute__((ext_vector_type(16)));
typedef float    v8f  __attribute__((ext_vector_type(8)));

namespace cfg {
constexpr int G    = 8;
constexpr int S    = 140;
constexpr int D    = 256;   // D_AR == D_ENC
constexpr int KP   = 12;    // prediction steps
constexpr int NEG  = 128;
constexpr int WIN  = 128;   // S - KP
constexpr int POOL = 20000;
constexpr int KSLOT = 16;   // KP padded to a full WMMA M-tile

constexpr int NW = KP * D * D;        // 786432  W elements
constexpr int NC = G * WIN * D;       // 262144  c elements
constexpr int NO = POOL * D;          // 5120000 otherEncoded elements
constexpr int NL = G * WIN * KSLOT * D; // 4194304 padded locC elements
}

// Load one WMMA 16-bit operand row per the CDNA5 16x16x32 A/B VGPR layout:
// lane = row(0..15) + 16*khalf ; v16h elems 0..7 <- K = kbase+khalf*8+0..7,
// elems 8..15 <- K = kbase+16+khalf*8+0..7. Two 16-byte loads.
__device__ inline v16h load_mat_row(const _Float16* row, int kbase, int khalf) {
  const v8h lo = *reinterpret_cast<const v8h*>(row + kbase + khalf * 8);
  const v8h hi = *reinterpret_cast<const v8h*>(row + kbase + 16 + khalf * 8);
  return __builtin_shufflevector(lo, hi, 0, 1, 2, 3, 4, 5, 6, 7,
                                 8, 9, 10, 11, 12, 13, 14, 15);
}

__device__ inline float wave_sum(float v) {
  for (int o = 16; o > 0; o >>= 1) v += __shfl_xor(v, o, 32);
  return v;
}
__device__ inline float wave_max(float v) {
  for (int o = 16; o > 0; o >>= 1) v = fmaxf(v, __shfl_xor(v, o, 32));
  return v;
}

// ---------------------------------------------------------------------------
// Kernel 0: zero the padded locC buffer (k-slots 12..15 must stay 0) and the
// 24 loss/acc accumulators.  Re-run every launch (graph-replay safe).
// ---------------------------------------------------------------------------
__global__ void k_zero(unsigned int* lz, float* accum) {
  const int n = cfg::NL / 2;  // halves -> u32 words
  for (int i = blockIdx.x * blockDim.x + threadIdx.x; i < n;
       i += gridDim.x * blockDim.x)
    lz[i] = 0u;
  if (blockIdx.x == 0 && threadIdx.x < 2 * cfg::KP) accum[threadIdx.x] = 0.f;
}

// Flat f32 -> f16 conversion (W, otherEncoded).
__global__ void k_cvt_flat(const float* __restrict__ src,
                           _Float16* __restrict__ dst, int n) {
  for (int i = blockIdx.x * blockDim.x + threadIdx.x; i < n;
       i += gridDim.x * blockDim.x)
    dst[i] = (_Float16)src[i];
}

// cFeature[:, :WIN, :] f32 -> f16 (drops the last K rows of S).
__global__ void k_cvt_c(const float* __restrict__ src,
                        _Float16* __restrict__ dst) {
  for (int i = blockIdx.x * blockDim.x + threadIdx.x; i < cfg::NC;
       i += gridDim.x * blockDim.x) {
    const int a = i & (cfg::D - 1);
    const int wp = (i / cfg::D) & (cfg::WIN - 1);
    const int g = i / (cfg::D * cfg::WIN);
    dst[i] = (_Float16)src[(g * cfg::S + wp) * cfg::D + a];
  }
}

// ---------------------------------------------------------------------------
// Kernel 1: locC[k,g,w,e] = sum_a c[g,w,a] * W[k,e,a]
// One wave computes one 16(w) x 16(e) f32 tile via 8x v_wmma_f32_16x16x32_f16.
// Output stored as Lh[g][w][kslot][e] f16 so stage 2 reads contiguous A rows.
// grid.x = KP * G * (WIN/16) * (D/16) = 12288 waves.
// ---------------------------------------------------------------------------
__global__ __launch_bounds__(32) void k_locc_gemm(
    const _Float16* __restrict__ Ch, const _Float16* __restrict__ Wh,
    _Float16* __restrict__ Lh) {
  const int bid = blockIdx.x;
  const int et = bid & 15;          // e tile (16 of them)
  const int wt = (bid >> 4) & 7;    // w tile (8)
  const int g  = (bid >> 7) & 7;    // group (8)
  const int k  = bid >> 10;         // prediction step (12)

  const int lane = threadIdx.x;
  const int lo = lane & 15;   // row within tile
  const int hi = lane >> 4;   // khalf

  const _Float16* arow = Ch + (g * cfg::WIN + wt * 16 + lo) * cfg::D;
  const _Float16* brow = Wh + (k * cfg::D + et * 16 + lo) * cfg::D;

  v8f acc = {};
#pragma unroll
  for (int kt = 0; kt < cfg::D / 32; ++kt) {
    const v16h a = load_mat_row(arow, kt * 32, hi);
    const v16h b = load_mat_row(brow, kt * 32, hi);
    acc = __builtin_amdgcn_wmma_f32_16x16x32_f16(false, a, false, b, (short)0,
                                                 acc, false, false);
  }

#pragma unroll
  for (int r = 0; r < 8; ++r) {
    const int m = r + 8 * hi;  // w within tile
    Lh[((g * cfg::WIN + wt * 16 + m) * cfg::KSLOT + k) * cfg::D + et * 16 + lo] =
        (_Float16)acc[r];
  }
}

// ---------------------------------------------------------------------------
// Kernel 2: per (g,w) position (1024 blocks, 1 wave each):
//   scoreNeg (12x128) = locC_gw (12x256, padded to 16) @ gathered negs^T,
//   via 8 N-tiles x 8 K-steps of v_wmma_f32_16x16x32_f16 -> LDS.
//   scorePos[k] = f32 dot(locC_gw[k], gt[g, w+k+1]) / 256 (VALU, wave-reduced)
//   129-way log-softmax per k; atomicAdd loss / acc accumulators.
// ---------------------------------------------------------------------------
__global__ __launch_bounds__(32) void k_score_softmax(
    const _Float16* __restrict__ Lh, const _Float16* __restrict__ Oh,
    const int* __restrict__ extIdx, const float* __restrict__ gt,
    float* __restrict__ accum) {
  __shared__ float sn[cfg::KSLOT * cfg::NEG];  // 8 KB of the 320 KB LDS

  const int b = blockIdx.x;
  const int g = b >> 7;
  const int wp = b & (cfg::WIN - 1);

  const int lane = threadIdx.x;
  const int lo = lane & 15;
  const int hi = lane >> 4;

  const _Float16* Abase = Lh + (g * cfg::WIN + wp) * cfg::KSLOT * cfg::D;
  const _Float16* arow = Abase + lo * cfg::D;

  // Preload the full 16x256 A tile (reused across all 8 N-tiles).
  v16h areg[cfg::D / 32];
#pragma unroll
  for (int kt = 0; kt < cfg::D / 32; ++kt)
    areg[kt] = load_mat_row(arow, kt * 32, hi);

  const int* idxBase = extIdx + (g * cfg::NEG) * cfg::WIN + wp;
  const float inv_d = 1.0f / (float)cfg::D;

  for (int nt = 0; nt < cfg::NEG / 16; ++nt) {
    const int n = nt * 16 + lo;
    const int id = idxBase[n * cfg::WIN];          // extIdx[g][n][wp]
    const _Float16* brow = Oh + id * cfg::D;       // gathered negative row

    v8f acc = {};
#pragma unroll
    for (int kt = 0; kt < cfg::D / 32; ++kt) {
      const v16h bmat = load_mat_row(brow, kt * 32, hi);
      acc = __builtin_amdgcn_wmma_f32_16x16x32_f16(false, areg[kt], false,
                                                   bmat, (short)0, acc, false,
                                                   false);
    }
#pragma unroll
    for (int r = 0; r < 8; ++r)
      sn[(r + 8 * hi) * cfg::NEG + nt * 16 + lo] = acc[r] * inv_d;
  }
  __syncthreads();

  // Per prediction step: positive score + log-softmax over 129 candidates.
  for (int k = 0; k < cfg::KP; ++k) {
    const _Float16* lrow = Abase + k * cfg::D;
    const float* prow = gt + (g * cfg::S + wp + k + 1) * cfg::D;
    float p = 0.f;
    for (int e = lane; e < cfg::D; e += 32) p += (float)lrow[e] * prow[e];
    const float sp = wave_sum(p) * inv_d;

    float mx = -1e30f;
#pragma unroll
    for (int j = 0; j < 4; ++j)
      mx = fmaxf(mx, sn[k * cfg::NEG + lane + 32 * j]);
    const float maxNeg = wave_max(mx);
    const float M = fmaxf(sp, maxNeg);

    float se = 0.f;
#pragma unroll
    for (int j = 0; j < 4; ++j)
      se += expf(sn[k * cfg::NEG + lane + 32 * j] - M);
    se = wave_sum(se) + expf(sp - M);

    const float logp0 = (sp - M) - logf(se);
    if (lane == 0) {
      atomicAdd(&accum[k], -logp0);
      atomicAdd(&accum[cfg::KP + k], (sp >= maxNeg) ? 1.0f : 0.0f);
    }
  }
}

// Kernel 3: scale accumulators by 1/(G*WIN) -> d_out = [losses(12) | acc(12)].
__global__ void k_finalize(const float* __restrict__ accum,
                           float* __restrict__ out) {
  const float inv = 1.0f / (float)(cfg::G * cfg::WIN);
  const int t = threadIdx.x;
  if (t < cfg::KP) {
    out[t] = accum[t] * inv;
    out[cfg::KP + t] = accum[cfg::KP + t] * inv;
  }
}

// ---------------------------------------------------------------------------
extern "C" void kernel_launch(void* const* d_in, const int* in_sizes, int n_in,
                              void* d_out, int out_size, void* d_ws,
                              size_t ws_size, hipStream_t stream) {
  const float* cFeature = (const float*)d_in[0];  // (G,S,D_AR)
  const float* gtPred   = (const float*)d_in[1];  // (G,S,D_ENC)
  const float* otherEnc = (const float*)d_in[2];  // (POOL,D_ENC)
  const int*   extIdx   = (const int*)d_in[3];    // (G,NEG,WIN)
  const float* Wmat     = (const float*)d_in[4];  // (K,D_ENC,D_AR)
  float* out = (float*)d_out;

  // Workspace carve-out (~19.8 MB): accum | Wh | Ch | Oh | Lh
  char* ws = (char*)d_ws;
  float*    accum = (float*)ws;                      // 24 f32
  _Float16* Wh = (_Float16*)(ws + 1024);             // NW halves
  _Float16* Ch = Wh + cfg::NW;                       // NC halves
  _Float16* Oh = Ch + cfg::NC;                       // NO halves
  _Float16* Lh = Oh + cfg::NO;                       // NL halves (k padded)

  k_zero<<<4096, 256, 0, stream>>>((unsigned int*)Lh, accum);
  k_cvt_flat<<<1024, 256, 0, stream>>>(Wmat, Wh, cfg::NW);
  k_cvt_c<<<512, 256, 0, stream>>>(cFeature, Ch);
  k_cvt_flat<<<4096, 256, 0, stream>>>(otherEnc, Oh, cfg::NO);

  const int gemm_blocks = cfg::KP * cfg::G * (cfg::WIN / 16) * (cfg::D / 16);
  k_locc_gemm<<<gemm_blocks, 32, 0, stream>>>(Ch, Wh, Lh);

  k_score_softmax<<<cfg::G * cfg::WIN, 32, 0, stream>>>(Lh, Oh, extIdx, gtPred,
                                                        accum);
  k_finalize<<<1, 32, 0, stream>>>(accum, out);
}
